// GSER_27865747817024
// MI455X (gfx1250) — matrix operand order repaired and checked by reading, not compiled, not run on hardware
//
#include <hip/hip_runtime.h>

// ---------------------------------------------------------------------------
// Fused gated-reservoir update for MI455X (gfx1250, wave32, WMMA).
//   out[:, :512] = spike(o * ((1-leak)*(f*prev) + leak*tanh(i*(X@WinT + prev@Wres))))
//   out[:, 512:] = 0
// All GEMMs run in f16 WMMA (16x16x32) with f32 accumulation, fused in one
// kernel; operands are pre-converted to f16 in workspace (~41 MB required).
// Phase-2 K loop is double-buffered; staging uses GLOBAL_LOAD_ASYNC_TO_LDS
// (ASYNCcnt-tracked) when the toolchain exposes the builtin, else ds_store.
// ---------------------------------------------------------------------------

typedef __attribute__((ext_vector_type(16))) _Float16 v16h;
typedef __attribute__((ext_vector_type(8)))  float    v8f;
typedef __attribute__((ext_vector_type(4)))  int      v4i;

#define BATCH   32768
#define IN_DIM  128
#define MAXD    1024
#define ACT     512

#define M_TILE  128
#define N_TILE  64
#define LDS_K   72        // padded LDS row stride (halfs) for a 64-wide K chunk

// phase-2 double-buffer byte offsets inside dynamic LDS
#define OFF_A0  0
#define OFF_A1  18432
#define OFF_B0  36864
#define OFF_B1  46080

// ---- workspace layout (bytes) ----
#define WS_X16      0                       // BATCH*128 f16   = 8,388,608
#define WS_P16      8388608                 // BATCH*512 f16   = 33,554,432
#define WS_WIN16    41943040                // 512*128 f16     = 131,072
#define WS_WREST16  42074112                // 512*512 f16     = 524,288
#define WS_WG16     42598400                // 1536*128 f16    = 393,216
// total ws requirement: 42,991,616 bytes

// ---- async global->LDS (gfx1250) with compile-safe fallback ----------------
#if __has_builtin(__builtin_amdgcn_global_load_async_to_lds_b128)
#define HAVE_ASYNC_LDS 1
typedef __attribute__((address_space(1))) v4i g1v4i;
typedef __attribute__((address_space(3))) v4i l3v4i;
__device__ __forceinline__ void copy16_async(void* lds_dst, const void* gsrc) {
  __builtin_amdgcn_global_load_async_to_lds_b128((g1v4i*)gsrc, (l3v4i*)lds_dst,
                                                 0, 0);
}
__device__ __forceinline__ void async_wait_all() {
#if __has_builtin(__builtin_amdgcn_s_wait_asynccnt)
  __builtin_amdgcn_s_wait_asynccnt(0);
#else
  asm volatile("s_wait_asynccnt 0x0" ::: "memory");
#endif
}
#define COPY16(dst, src) copy16_async((dst), (src))
#else
#define HAVE_ASYNC_LDS 0
__device__ __forceinline__ void async_wait_all() {}
#define COPY16(dst, src) (*(uint4*)(dst) = *(const uint4*)(src))
#endif

union Frag16 { v16h v; uint4 u[2]; };

// A-matrix 16x32 f16 fragment (ISA 7.12.2): lane<16 -> row=lane, K={0..7,16..23};
// lane>=16 -> row=lane-16, K={8..15,24..31}.
__device__ __forceinline__ void load_a_frag(Frag16& f, const _Float16* m,
                                            int rowBase, int k0, int ld, int lane) {
  int r  = rowBase + (lane & 15);
  int kl = k0 + (lane >> 4) * 8;
  f.u[0] = *(const uint4*)(m + r * ld + kl);
  f.u[1] = *(const uint4*)(m + r * ld + kl + 16);
}

// B-matrix 32x16 f16 fragment: lanes 0-15 hold K={0..7,16..23} of col=lane,
// lanes 16-31 hold K={8..15,24..31} of col=lane-16. Storage is col-major
// (output-col rows, K contiguous), so each 8-half run is contiguous.
__device__ __forceinline__ void load_b_frag(Frag16& f, const _Float16* m,
                                            int rowBase, int k0, int ld, int lane) {
  int r  = rowBase + (lane & 15);
  int ks = k0 + (lane >> 4) * 8;
  f.u[0] = *(const uint4*)(m + r * ld + ks);
  f.u[1] = *(const uint4*)(m + r * ld + ks + 16);
}

__device__ __forceinline__ v8f wmma_f16(const Frag16& a, const Frag16& b, v8f c) {
  return __builtin_amdgcn_wmma_f32_16x16x32_f16(false, a.v, false, b.v,
                                                (short)0, c, false, false);
}

__device__ __forceinline__ float sigmoidf(float x) {
  return 1.0f / (1.0f + __expf(-x));
}

// ---------------------------------------------------------------------------
// prep kernels
// ---------------------------------------------------------------------------
__global__ __launch_bounds__(256) void k_cvt_copy(const float* __restrict__ src,
                                                  _Float16* __restrict__ dst, int n) {
  int i = blockIdx.x * 256 + threadIdx.x;
  if (i < n) dst[i] = (_Float16)src[i];
}

__global__ __launch_bounds__(256) void k_cvt_prev(const float* __restrict__ prev,
                                                  _Float16* __restrict__ dst) {
  int i = blockIdx.x * 256 + threadIdx.x;   // i = b*512 + k, k < 512
  int b = i >> 9, k = i & 511;
  dst[i] = (_Float16)prev[(size_t)b * MAXD + k];
}

__global__ __launch_bounds__(256) void k_cvt_wrest(const float* __restrict__ W_res,
                                                   _Float16* __restrict__ dst) {
  int i = blockIdx.x * 256 + threadIdx.x;   // i = n*512 + k
  int n = i >> 9, k = i & 511;
  dst[i] = (_Float16)W_res[(size_t)k * MAXD + n];   // transpose: [n][k] <- W_res[k][n]
}

__global__ __launch_bounds__(256) void k_pad_zero(float* __restrict__ out) {
  int i = blockIdx.x * 256 + threadIdx.x;   // i = row*512 + c
  int row = i >> 9, c = i & 511;
  out[(size_t)row * MAXD + ACT + c] = 0.0f;
}

// ---------------------------------------------------------------------------
// staging helpers (issue COPY16s; completion via async_wait_all + barrier)
// ---------------------------------------------------------------------------
__device__ __forceinline__ void stage_phase1(_Float16* sA, _Float16* sB,
    const _Float16* X16, const _Float16* Win16, const _Float16* Wg16,
    int mBase, int nBase, int k0, int tid) {
  for (int i = tid; i < 128 * 8; i += 256) {        // X tile: 128 x 64 halfs
    int r = i >> 3, c = (i & 7) * 8;
    COPY16(sA + r * LDS_K + c, X16 + (size_t)(mBase + r) * IN_DIM + k0 + c);
  }
  for (int i = tid; i < 256 * 8; i += 256) {        // 4 weight slabs: 256 x 64
    int r = i >> 3, c = (i & 7) * 8;
    const _Float16* src;
    if (r < 64) {
      src = Win16 + (size_t)(nBase + r) * IN_DIM;
    } else {
      int g = (r - 64) >> 6, rr = (r - 64) & 63;    // gate slab 0..2
      src = Wg16 + (size_t)(g * ACT + nBase + rr) * IN_DIM;
    }
    COPY16(sB + r * LDS_K + c, src + k0 + c);
  }
}

__device__ __forceinline__ void stage_phase2(_Float16* sA, _Float16* sB,
    const _Float16* P16, const _Float16* WresT16,
    int mBase, int nBase, int k0, int tid) {
  for (int i = tid; i < 128 * 8; i += 256) {        // prev tile: 128 x 64
    int r = i >> 3, c = (i & 7) * 8;
    COPY16(sA + r * LDS_K + c, P16 + (size_t)(mBase + r) * ACT + k0 + c);
  }
  for (int i = tid; i < 64 * 8; i += 256) {         // W_res^T tile: 64 x 64
    int r = i >> 3, c = (i & 7) * 8;
    COPY16(sB + r * LDS_K + c, WresT16 + (size_t)(nBase + r) * ACT + k0 + c);
  }
}

// ---------------------------------------------------------------------------
// fused GEMM + epilogue
//   grid = (ACT/N_TILE, BATCH/M_TILE) = (8, 256), block = 256 threads = 8 waves
//   wave (wm, wn) in 4x2 owns a 32x32 patch = 2x2 wmma tiles x 4 accumulators
// ---------------------------------------------------------------------------
__global__ __launch_bounds__(256) void k_fused(
    const _Float16* __restrict__ X16, const _Float16* __restrict__ P16,
    const _Float16* __restrict__ Win16, const _Float16* __restrict__ WresT16,
    const _Float16* __restrict__ Wg16,
    const float* __restrict__ prev_f32,
    const float* __restrict__ leak_p, const float* __restrict__ thr_p,
    float* __restrict__ out)
{
  extern __shared__ char smem[];
  // Phase 1: sA = [0,18432), sB = [18432,55296)
  _Float16* sA1 = (_Float16*)smem;
  _Float16* sB1 = (_Float16*)(smem + 128 * LDS_K * 2);

  const int tid   = threadIdx.x;
  const int lane  = tid & 31;
  const int wid   = tid >> 5;
  const int wm    = wid & 3;             // 0..3 (M)
  const int wn    = wid >> 2;            // 0..1 (N)
  const int mBase = blockIdx.y * M_TILE;
  const int nBase = blockIdx.x * N_TILE;

  v8f accM[2][2] = {};
  v8f accI[2][2] = {};
  v8f accF[2][2] = {};
  v8f accO[2][2] = {};

  // ---------- Phase 1: K = 128 against X (W_in + 3 gate slabs share A) ------
  #pragma unroll 1
  for (int k0 = 0; k0 < IN_DIM; k0 += 64) {
    __syncthreads();                       // prior readers of sA1/sB1 done
    stage_phase1(sA1, sB1, X16, Win16, Wg16, mBase, nBase, k0, tid);
    async_wait_all();
    __syncthreads();                       // staged data visible to all waves

    #pragma unroll
    for (int kk = 0; kk < 64; kk += 32) {
      Frag16 a[2];
      #pragma unroll
      for (int mi = 0; mi < 2; ++mi)
        load_a_frag(a[mi], sA1, wm * 32 + mi * 16, kk, LDS_K, lane);
      #pragma unroll
      for (int ni = 0; ni < 2; ++ni) {
        const int cr = wn * 32 + ni * 16;
        Frag16 bw, bi, bf, bo;
        load_b_frag(bw, sB1,        cr, kk, LDS_K, lane);
        load_b_frag(bi, sB1,  64 + cr, kk, LDS_K, lane);
        load_b_frag(bf, sB1, 128 + cr, kk, LDS_K, lane);
        load_b_frag(bo, sB1, 192 + cr, kk, LDS_K, lane);
        #pragma unroll
        for (int mi = 0; mi < 2; ++mi) {
          accM[mi][ni] = wmma_f16(a[mi], bw, accM[mi][ni]);
          accI[mi][ni] = wmma_f16(a[mi], bi, accI[mi][ni]);
          accF[mi][ni] = wmma_f16(a[mi], bf, accF[mi][ni]);
          accO[mi][ni] = wmma_f16(a[mi], bo, accO[mi][ni]);
        }
      }
    }
  }

  // ---------- Phase 2: K = 512 reservoir GEMM into accM (double buffered) ----
  __syncthreads();                         // all waves done reading phase-1 LDS
  stage_phase2((_Float16*)(smem + OFF_A0), (_Float16*)(smem + OFF_B0),
               P16, WresT16, mBase, nBase, 0, tid);

  #pragma unroll 1
  for (int it = 0; it < ACT / 64; ++it) {
    const int cur = it & 1;
    _Float16* cA = (_Float16*)(smem + (cur ? OFF_A1 : OFF_A0));
    _Float16* cB = (_Float16*)(smem + (cur ? OFF_B1 : OFF_B0));

    async_wait_all();                      // our async writes for chunk `it` done
    __syncthreads();                       // chunk `it` visible; other buf free
    if (it + 1 < ACT / 64) {               // prefetch next chunk under compute
      _Float16* nA = (_Float16*)(smem + (cur ? OFF_A0 : OFF_A1));
      _Float16* nB = (_Float16*)(smem + (cur ? OFF_B0 : OFF_B1));
      stage_phase2(nA, nB, P16, WresT16, mBase, nBase, (it + 1) * 64, tid);
    }

    #pragma unroll
    for (int kk = 0; kk < 64; kk += 32) {
      Frag16 a[2];
      #pragma unroll
      for (int mi = 0; mi < 2; ++mi)
        load_a_frag(a[mi], cA, wm * 32 + mi * 16, kk, LDS_K, lane);
      #pragma unroll
      for (int ni = 0; ni < 2; ++ni) {
        Frag16 b;
        load_b_frag(b, cB, wn * 32 + ni * 16, kk, LDS_K, lane);
        #pragma unroll
        for (int mi = 0; mi < 2; ++mi)
          accM[mi][ni] = wmma_f16(a[mi], b, accM[mi][ni]);
      }
    }
  }

  // ---------- Epilogue (fp32) ------------------------------------------------
  // C/D layout: lane<16 -> N=lane, M=v; lane>=16 -> N=lane-16, M=v+8.
  #pragma unroll
  for (int ni = 0; ni < 2; ++ni) {
    const int col  = nBase + wn * 32 + ni * 16 + (lane & 15);
    const float leak = sigmoidf(leak_p[col]);
    const float thr  = log1pf(__expf(thr_p[col]));        // softplus
    #pragma unroll
    for (int mi = 0; mi < 2; ++mi) {
      const int row0 = mBase + wm * 32 + mi * 16 + (lane >> 4) * 8;
      #pragma unroll
      for (int v = 0; v < 8; ++v) {
        const int row = row0 + v;
        const float gi = sigmoidf(accI[mi][ni][v]);
        const float gf = sigmoidf(accF[mi][ni][v]);
        const float go = sigmoidf(accO[mi][ni][v]);
        const float pv = prev_f32[(size_t)row * MAXD + col];
        float st = (1.0f - leak) * (gf * pv) + leak * tanhf(gi * accM[mi][ni][v]);
        st *= go;
        st = (st > thr) ? (st - thr) : st;
        out[(size_t)row * MAXD + col] = st;
      }
    }
  }
}

// ---------------------------------------------------------------------------
extern "C" void kernel_launch(void* const* d_in, const int* in_sizes, int n_in,
                              void* d_out, int out_size, void* d_ws, size_t ws_size,
                              hipStream_t stream) {
  const float* inputs = (const float*)d_in[0];   // (32768, 128)
  const float* prev   = (const float*)d_in[1];   // (32768, 1024)
  const float* W_res  = (const float*)d_in[2];   // (1024, 1024)
  const float* W_in   = (const float*)d_in[3];   // (1024, 128)
  const float* W_gate = (const float*)d_in[4];   // (3072, 128)
  const float* leak_p = (const float*)d_in[5];   // (1024,)
  const float* thr_p  = (const float*)d_in[6];   // (1024,)
  float* out = (float*)d_out;                    // (32768, 1024)
  (void)in_sizes; (void)n_in; (void)out_size; (void)ws_size;

  char* ws = (char*)d_ws;
  _Float16* X16     = (_Float16*)(ws + WS_X16);
  _Float16* P16     = (_Float16*)(ws + WS_P16);
  _Float16* Win16   = (_Float16*)(ws + WS_WIN16);
  _Float16* WresT16 = (_Float16*)(ws + WS_WREST16);
  _Float16* Wg16    = (_Float16*)(ws + WS_WG16);

  // f32 -> f16 operand prep
  k_cvt_copy <<<(BATCH * IN_DIM) / 256, 256, 0, stream>>>(inputs, X16, BATCH * IN_DIM);
  k_cvt_prev <<<(BATCH * ACT) / 256, 256, 0, stream>>>(prev, P16);
  k_cvt_copy <<<(ACT * IN_DIM) / 256, 256, 0, stream>>>(W_in, Win16, ACT * IN_DIM);
  k_cvt_copy <<<(3 * ACT * IN_DIM) / 256, 256, 0, stream>>>(W_gate, Wg16, 3 * ACT * IN_DIM);
  k_cvt_wrest<<<(ACT * ACT) / 256, 256, 0, stream>>>(W_res, WresT16);

  // zero the padded columns [512, 1024)
  k_pad_zero<<<(BATCH * (MAXD - ACT)) / 256, 256, 0, stream>>>(out);

  // fused GEMMs + epilogue
  dim3 grid(ACT / N_TILE, BATCH / M_TILE);                 // (8, 256)
  size_t smem = (size_t)(128 + 256) * LDS_K * sizeof(_Float16);  // 55,296 B
  k_fused<<<grid, 256, smem, stream>>>(X16, P16, Win16, WresT16, Wg16,
                                       prev, leak_p, thr_p, out);
}
